// GNNPolicy_75127567942264
// MI455X (gfx1250) — matrix operand order repaired and checked by reading, not compiled, not run on hardware
//
#include <hip/hip_runtime.h>
#include <math.h>

#define NCONS 100000
#define NVARS 100000
#define NGRAPH 64

typedef __attribute__((ext_vector_type(16))) _Float16 v16h;
typedef __attribute__((ext_vector_type(8)))  float    v8f;
typedef __attribute__((ext_vector_type(2)))  _Float16 h2;

// ---------------- WMMA helpers (CDNA5 wave32, 16x16x32 f16 -> f32) ----------------

__device__ __forceinline__ v8f wmma_f16(v16h a, v16h b, v8f c) {
  // 8 args: (neg_a, A, neg_b, B, c_mod, C, reuse_a, reuse_b)
  return __builtin_amdgcn_wmma_f32_16x16x32_f16(false, a, false, b, (short)0, c,
                                                false, false);
}

// A-matrix 16x32 f16 fragment from LDS row-major tile:
// lane<16: M=lane, K = kOff+{0..15}
// lane>=16: M=lane-16, K = kOff+{16..31}
__device__ __forceinline__ v16h ldsA(const _Float16* base, int rowStride, int kOff, int lane) {
  int row = lane & 15;
  int hi  = (lane >> 4) << 4;  // 0 or 16 halves
  const _Float16* p0 = base + row * rowStride + kOff + hi;
  v16h a;
  *(uint4*)&a         = *(const uint4*)p0;
  *(((uint4*)&a) + 1) = *(const uint4*)(p0 + 8);
  return a;
}

// B-matrix 32x16 f16 fragment, pre-swizzled in global: frag[lane][16 halves]
// (lane<16: N=lane, K=0..15; lane>=16: N=lane-16, K=16..31)
__device__ __forceinline__ v16h ldB(const _Float16* frag, int lane) {
  const _Float16* p = frag + lane * 16;
  v16h b;
  *(uint4*)&b         = *(const uint4*)p;
  *(((uint4*)&b) + 1) = *(const uint4*)(p + 8);
  return b;
}

// ---------------- conv column mapping (HIDDEN_DIMS = 8,8,8,1 -> 25 cols) ----------

struct ConvP { const float *eW, *eb, *mW, *mb, *sW, *sb; int d; };
struct ConvAll { ConvP c[4]; };

__device__ __forceinline__ void conv_of(int col, int& ci, int& cj) {
  if (col < 8)       { ci = 0; cj = col; }
  else if (col < 16) { ci = 1; cj = col - 8; }
  else if (col < 24) { ci = 2; cj = col - 16; }
  else               { ci = 3; cj = col - 24; }
}

// ---------------- constants: LN over size-1 axis is a parameter constant ----------

__global__ void const_prep(ConvAll cp,
                           const float* c_l1W, const float* c_l1b,
                           const float* c_l2W, const float* c_l2b,
                           const float* c_lnb, const float* e_lnb,
                           h2* edge_add2, float* var_const, float* cons_self,
                           float* bc_cat) {
  __shared__ float hC[64], cC[64], sEm[26];
  int t = threadIdx.x;
  float lnc = c_lnb[0];                       // LN(cons) == cons_ln.b (constant)
  hC[t] = fmaxf(lnc * c_l1W[t] + c_l1b[t], 0.f);
  __syncthreads();
  float acc = c_l2b[t];
  for (int k = 0; k < 64; k++) acc += hC[k] * c_l2W[k * 64 + t];
  cC[t] = fmaxf(acc, 0.f);                    // constant c-embedding row
  __syncthreads();
  if (t < 25) {
    int ci, cj; conv_of(t, ci, cj);
    ConvP p = cp.c[ci];
    float lne = e_lnb[0];                     // LN(ef) == edge_ln.b (constant)
    float em = lne * p.eW[cj] + p.eb[cj];
    float cm = p.mb[cj], cs = p.sb[cj];
    for (int k = 0; k < 64; k++) {
      cm += cC[k] * p.mW[k * p.d + cj];
      cs += cC[k] * p.sW[k * p.d + cj];
    }
    sEm[t]       = em;           // added per edge into agg_c
    var_const[t] = cm + em;      // agg_v[i] = deg_src[i] * var_const
    cons_self[t] = cs;           // constant constraint self-term
    bc_cat[t]      = p.mb[cj];   // fused bias for [64x50] projection
    bc_cat[25 + t] = p.sb[cj];
  }
  if (t == 25) sEm[25] = 0.f;
  if (t >= 50) bc_cat[t] = 0.f;  // pad cols 50..63
  __syncthreads();
  if (t < 13) {                  // pre-pack edge constant as half2 for pk atomics
    h2 v;
    v[0] = (_Float16)sEm[2 * t];
    v[1] = (_Float16)sEm[2 * t + 1];
    edge_add2[t] = v;
  }
}

// ---------------- B-fragment pre-swizzle for the three GEMMs ----------------------

__global__ void frag_prep(const float* W1, const float* W2, ConvAll cp,
                          _Float16* fW1, _Float16* fW2, _Float16* fWc) {
  int tid = blockIdx.x * blockDim.x + threadIdx.x;
  if (tid >= 20 * 512) return;               // 4 + 8 + 8 fragments of 32*16 halves
  int frag = tid >> 9, rem = tid & 511;
  int lane = rem >> 4, h = rem & 15;
  int nl = lane & 15;
  int kc32 = ((lane >> 4) << 4) | h;         // k within a 32-chunk
  if (frag < 4) {                            // W1 padded 32x64 (rows 6..31 = 0)
    int n = frag * 16 + nl, k = kc32;
    fW1[tid] = (_Float16)((k < 6) ? W1[k * 64 + n] : 0.f);
  } else if (frag < 12) {                    // W2 64x64: frag = tile*2 + kchunk
    int f = frag - 4, t = f >> 1, kc = f & 1;
    int n = t * 16 + nl, k = kc * 32 + kc32;
    fW2[f * 512 + lane * 16 + h] = (_Float16)W2[k * 64 + n];
  } else {                                   // Wc 64x64 = [msg_cat | self_cat | 0]
    int f = frag - 12, t = f >> 1, kc = f & 1;
    int n = t * 16 + nl, k = kc * 32 + kc32;
    float val = 0.f;
    if (n < 25)      { int ci, cj; conv_of(n, ci, cj);      val = cp.c[ci].mW[k * cp.c[ci].d + cj]; }
    else if (n < 50) { int ci, cj; conv_of(n - 25, ci, cj); val = cp.c[ci].sW[k * cp.c[ci].d + cj]; }
    fWc[f * 512 + lane * 16 + h] = (_Float16)val;
  }
}

// ---------------- fused variable path: LN -> 3 WMMA GEMM stages -------------------

__global__ __launch_bounds__(256) void var_embed_kernel(
    const float* __restrict__ var, const float* lng, const float* lnb,
    const float* b1, const float* b2, const float* bc,
    const _Float16* __restrict__ fW1, const _Float16* __restrict__ fW2,
    const _Float16* __restrict__ fWc,
    _Float16* __restrict__ vm, float* __restrict__ sv) {
  // per-wave tile buffers: x[16x32] h[16x64] v[16x64] = 2560 halves
  __shared__ __align__(16) _Float16 smem[8 * 2560];
  int wave = threadIdx.x >> 5, lane = threadIdx.x & 31;
  _Float16* xb = smem + wave * 2560;
  _Float16* hb = xb + 16 * 32;
  _Float16* vb = hb + 16 * 64;
  int rbase = (blockIdx.x * 8 + wave) * 16;
  bool act = rbase < NVARS;
  int nl = lane & 15, hi8 = (lane >> 4) * 8;

  // stage 0: per-row layernorm over 6 features, zero-pad K to 32
  if (act && lane < 16) {
    const float* xr = var + (size_t)(rbase + lane) * 6;
    float x0[6]; float m = 0.f;
    for (int j = 0; j < 6; j++) { x0[j] = xr[j]; m += x0[j]; }
    m *= (1.f / 6.f);
    float vv = 0.f;
    for (int j = 0; j < 6; j++) { float d = x0[j] - m; vv += d * d; }
    vv *= (1.f / 6.f);
    float inv = rsqrtf(vv + 1e-5f);
    _Float16* row = xb + lane * 32;
    for (int j = 0; j < 6; j++)  row[j] = (_Float16)((x0[j] - m) * inv * lng[j] + lnb[j]);
    for (int j = 6; j < 32; j++) row[j] = (_Float16)0.f;
  }
  __syncthreads();

  // stage 1: h = relu(x @ W1 + b1)   [16x32]*[32x64]
  if (act) {
    v16h a = ldsA(xb, 32, 0, lane);
    for (int t = 0; t < 4; t++) {
      int col = t * 16 + nl;
      float bias = b1[col];
      v8f c; for (int r = 0; r < 8; r++) c[r] = bias;
      c = wmma_f16(a, ldB(fW1 + t * 512, lane), c);
      for (int r = 0; r < 8; r++) hb[(r + hi8) * 64 + col] = (_Float16)fmaxf(c[r], 0.f);
    }
  }
  __syncthreads();

  // stage 2: v = relu(h @ W2 + b2)   [16x64]*[64x64], 2 K-steps
  if (act) {
    v16h a0 = ldsA(hb, 64, 0, lane), a1 = ldsA(hb, 64, 32, lane);
    for (int t = 0; t < 4; t++) {
      int col = t * 16 + nl;
      float bias = b2[col];
      v8f c; for (int r = 0; r < 8; r++) c[r] = bias;
      c = wmma_f16(a0, ldB(fW2 + (t * 2 + 0) * 512, lane), c);
      c = wmma_f16(a1, ldB(fW2 + (t * 2 + 1) * 512, lane), c);
      for (int r = 0; r < 8; r++) vb[(r + hi8) * 64 + col] = (_Float16)fmaxf(c[r], 0.f);
    }
  }
  __syncthreads();

  // stage 3: [vm | sv] = v @ Wc + bc  (cols 0..24 -> vm f16, 25..49 -> sv f32)
  if (act) {
    v16h a0 = ldsA(vb, 64, 0, lane), a1 = ldsA(vb, 64, 32, lane);
    for (int t = 0; t < 4; t++) {
      int col = t * 16 + nl;
      float bias = bc[col];
      v8f c; for (int r = 0; r < 8; r++) c[r] = bias;
      c = wmma_f16(a0, ldB(fWc + (t * 2 + 0) * 512, lane), c);
      c = wmma_f16(a1, ldB(fWc + (t * 2 + 1) * 512, lane), c);
      if (col < 25) {
        for (int r = 0; r < 8; r++)
          vm[(size_t)(rbase + r + hi8) * 32 + col] = (_Float16)c[r];
      } else if (col < 50) {
        for (int r = 0; r < 8; r++)
          sv[(size_t)(rbase + r + hi8) * 32 + (col - 25)] = c[r];
        if (col < 32) {  // keep vm pad halves (25..31) zero for packed atomics
          for (int r = 0; r < 8; r++)
            vm[(size_t)(rbase + r + hi8) * 32 + col] = (_Float16)0.f;
        }
      }
    }
  }
}

// ---------------- edge scatter: agg_c[dst] += vm[src] + emsg; deg_src[src]++ ------
// 13 packed f16 atomics per edge (GLOBAL_ATOMIC_PK_ADD_F16) instead of 25 f32.

__global__ __launch_bounds__(256) void edge_kernel(
    const int* __restrict__ ei, int ne,
    const _Float16* __restrict__ vm, const h2* __restrict__ edge_add2,
    _Float16* __restrict__ agg, int* __restrict__ deg) {
  int e = blockIdx.x * blockDim.x + threadIdx.x;
  if (e >= ne) return;
  int s = ei[e], d = ei[ne + e];
  atomicAdd(&deg[s], 1);
  const h2* vr = (const h2*)(vm + (size_t)s * 32);
  _Float16* ar = agg + (size_t)d * 32;
#pragma unroll
  for (int p = 0; p < 13; p++) {
    h2 sum = vr[p] + edge_add2[p];          // v_pk_add_f16
    unsigned long long addr = (unsigned long long)(size_t)(ar + p * 2);
    asm volatile("global_atomic_pk_add_f16 %0, %1, off"
                 :: "v"(addr), "v"(sum) : "memory");
  }
}

// ---------------- segment-mean pooling (sorted batch -> LDS accumulate) -----------

__global__ __launch_bounds__(256) void var_pool_kernel(
    const float* __restrict__ sv, const int* __restrict__ deg,
    const int* __restrict__ vb, const float* __restrict__ var_const,
    float* __restrict__ vsum) {
  __shared__ float ls[64 * 26];
  for (int i = threadIdx.x; i < 64 * 26; i += 256) ls[i] = 0.f;
  __syncthreads();
  int i = blockIdx.x * 256 + threadIdx.x;
  if (i < NVARS) {
    int b = vb[i];
    float dg = (float)deg[i];
    const float* sr = sv + (size_t)i * 32;
    for (int k = 0; k < 25; k++)
      atomicAdd(&ls[b * 26 + k], fmaxf(dg * var_const[k] + sr[k], 0.f));
    atomicAdd(&ls[b * 26 + 25], 1.f);
  }
  __syncthreads();
  int t = threadIdx.x;
  if (t < 64) {
    float cnt = ls[t * 26 + 25];
    if (cnt > 0.f) {
      for (int k = 0; k < 25; k++) atomicAdd(&vsum[t * 26 + k], ls[t * 26 + k]);
      atomicAdd(&vsum[t * 26 + 25], cnt);
    }
  }
}

__global__ __launch_bounds__(256) void cons_pool_kernel(
    const _Float16* __restrict__ agg, const int* __restrict__ cb,
    const float* __restrict__ cons_self, float* __restrict__ csum) {
  __shared__ float ls[64 * 26];
  for (int i = threadIdx.x; i < 64 * 26; i += 256) ls[i] = 0.f;
  __syncthreads();
  int j = blockIdx.x * 256 + threadIdx.x;
  if (j < NCONS) {
    int b = cb[j];
    const _Float16* ar = agg + (size_t)j * 32;
    for (int k = 0; k < 25; k++)
      atomicAdd(&ls[b * 26 + k], fmaxf((float)ar[k] + cons_self[k], 0.f));
    atomicAdd(&ls[b * 26 + 25], 1.f);
  }
  __syncthreads();
  int t = threadIdx.x;
  if (t < 64) {
    float cnt = ls[t * 26 + 25];
    if (cnt > 0.f) {
      for (int k = 0; k < 25; k++) atomicAdd(&csum[t * 26 + k], ls[t * 26 + k]);
      atomicAdd(&csum[t * 26 + 25], cnt);
    }
  }
}

__global__ void pool_final(const float* vsum, const float* csum, float* s) {
  int idx = blockIdx.x * blockDim.x + threadIdx.x;
  if (idx >= 64 * 50) return;
  int r = idx / 50, k = idx % 50;
  float val = (k < 25) ? vsum[r * 26 + k] / fmaxf(vsum[r * 26 + 25], 1.f)
                       : csum[r * 26 + (k - 25)] / fmaxf(csum[r * 26 + 25], 1.f);
  s[r * 50 + k] = val;
}

// ---------------- head: LN(s1-s0) -> 50x256 relu -> 256x1 -> sigmoid --------------

__global__ __launch_bounds__(256) void head_kernel(
    const float* s0, const float* s1,
    const float* lng, const float* lnb,
    const float* W1, const float* b1, const float* W2, float* out) {
  __shared__ float xs[50];
  __shared__ float red[256];
  int row = blockIdx.x, t = threadIdx.x;
  if (t == 0) {
    float d[50]; float m = 0.f;
    for (int k = 0; k < 50; k++) { d[k] = s1[row * 50 + k] - s0[row * 50 + k]; m += d[k]; }
    m *= 0.02f;
    float v = 0.f;
    for (int k = 0; k < 50; k++) { float q = d[k] - m; v += q * q; }
    v *= 0.02f;
    float inv = rsqrtf(v + 1e-5f);
    for (int k = 0; k < 50; k++) xs[k] = (d[k] - m) * inv * lng[k] + lnb[k];
  }
  __syncthreads();
  float h = b1[t];
  for (int k = 0; k < 50; k++) h += xs[k] * W1[k * 256 + t];
  h = fmaxf(h, 0.f);
  red[t] = h * W2[t];
  __syncthreads();
  for (int s = 128; s > 0; s >>= 1) {
    if (t < s) red[t] += red[t + s];
    __syncthreads();
  }
  if (t == 0) out[row] = 1.f / (1.f + expf(-red[0]));
}

// ---------------- host orchestration ----------------------------------------------
// Param leaf order (jax tree flatten, alphabetical keys, 'W' < 'b'):
// 0-1 cons_l1.{W,b} 2-3 cons_l2.{W,b} 4-5 cons_ln.{b,g}
// 6..29 convs[i].{edge.W,edge.b,msg.W,msg.b,self.W,self.b} i=0..3
// 30-31 edge_ln.{b,g} 32-33 final_l1.{W,b} 34 final_l2.W
// 35-36 final_ln.{b,g} 37-38 var_l1.{W,b} 39-40 var_l2.{W,b} 41-42 var_ln.{b,g}
// then per graph g: 43+6g: cons, ei, ef, var, cbatch, vbatch

extern "C" void kernel_launch(void* const* d_in, const int* in_sizes, int n_in,
                              void* d_out, int out_size, void* d_ws, size_t ws_size,
                              hipStream_t stream) {
  (void)n_in; (void)out_size; (void)ws_size;
  const float* P[43];
  for (int i = 0; i < 43; i++) P[i] = (const float*)d_in[i];
  static const int dims[4] = {8, 8, 8, 1};
  ConvAll cp;
  for (int i = 0; i < 4; i++) {
    int b = 6 + 6 * i;
    cp.c[i].eW = P[b];     cp.c[i].eb = P[b + 1];
    cp.c[i].mW = P[b + 2]; cp.c[i].mb = P[b + 3];
    cp.c[i].sW = P[b + 4]; cp.c[i].sb = P[b + 5];
    cp.c[i].d = dims[i];
  }

  char* ws = (char*)d_ws;
  h2*       edge_add2 = (h2*)(ws + 0);
  float*    var_const = (float*)(ws + 128);
  float*    cons_self = (float*)(ws + 256);
  float*    bc_cat    = (float*)(ws + 384);
  _Float16* fW1 = (_Float16*)(ws + 1024);
  _Float16* fW2 = (_Float16*)(ws + 1024 + 4096);
  _Float16* fWc = (_Float16*)(ws + 1024 + 4096 + 8192);
  size_t o = 32768;
  _Float16* vm  = (_Float16*)(ws + o); o += (size_t)NVARS * 32 * 2;
  float*    sv  = (float*)(ws + o);    o += (size_t)NVARS * 32 * 4;
  _Float16* agg = (_Float16*)(ws + o); o += (size_t)NCONS * 32 * 2;
  int*      deg = (int*)(ws + o);      o += (size_t)NVARS * 4;
  float*    vsum = (float*)(ws + o);   o += 64 * 26 * 4;
  float*    csum = (float*)(ws + o);   o += 64 * 26 * 4;
  float*    s0   = (float*)(ws + o);   o += 64 * 50 * 4;
  float*    s1   = (float*)(ws + o);

  hipLaunchKernelGGL(const_prep, dim3(1), dim3(64), 0, stream, cp,
                     P[0], P[1], P[2], P[3], P[4], P[30],
                     edge_add2, var_const, cons_self, bc_cat);
  hipLaunchKernelGGL(frag_prep, dim3(40), dim3(256), 0, stream,
                     P[37], P[39], cp, fW1, fW2, fWc);

  for (int g = 0; g < 2; g++) {
    int base = 43 + g * 6;
    const int*   ei  = (const int*)d_in[base + 1];
    int          ne  = in_sizes[base + 1] / 2;
    const float* var = (const float*)d_in[base + 3];
    const int*   cb  = (const int*)d_in[base + 4];
    const int*   vb  = (const int*)d_in[base + 5];
    float* sg = g ? s1 : s0;

    hipMemsetAsync(agg, 0, (size_t)NCONS * 32 * 2, stream);   // f16 +0
    hipMemsetAsync(deg, 0, (size_t)NVARS * 4, stream);
    hipMemsetAsync(vsum, 0, (size_t)2 * 64 * 26 * 4, stream); // vsum+csum contiguous

    hipLaunchKernelGGL(var_embed_kernel, dim3((NVARS / 16 + 7) / 8), dim3(256), 0, stream,
                       var, P[42], P[41], P[38], P[40], bc_cat, fW1, fW2, fWc, vm, sv);
    hipLaunchKernelGGL(edge_kernel, dim3((ne + 255) / 256), dim3(256), 0, stream,
                       ei, ne, vm, edge_add2, agg, deg);
    hipLaunchKernelGGL(var_pool_kernel, dim3((NVARS + 255) / 256), dim3(256), 0, stream,
                       sv, deg, vb, var_const, vsum);
    hipLaunchKernelGGL(cons_pool_kernel, dim3((NCONS + 255) / 256), dim3(256), 0, stream,
                       agg, cb, cons_self, csum);
    hipLaunchKernelGGL(pool_final, dim3((64 * 50 + 255) / 256), dim3(256), 0, stream,
                       vsum, csum, sg);
  }

  hipLaunchKernelGGL(head_kernel, dim3(64), dim3(256), 0, stream,
                     s0, s1, P[36], P[35], P[32], P[33], P[34], (float*)d_out);
}
